// MultiHeadAttention_89111981458027
// MI455X (gfx1250) — matrix operand order repaired and checked by reading, compiled, not run on hardware
//
#include <hip/hip_runtime.h>
#include <hip/hip_bf16.h>

// Problem constants (fixed by the reference)
#define B_  2
#define S_  4096
#define D_  512
#define H_  8
#define HD_ 64
#define M_  (B_ * S_)          // 8192 rows of x / attn output
#define SCALE_ 0.125f          // 1/sqrt(HD)

typedef __attribute__((ext_vector_type(16))) _Float16 v16h;
typedef __attribute__((ext_vector_type(8)))  _Float16 v8h;
typedef __attribute__((ext_vector_type(8)))  float    v8f;

union Frag16 {
    v16h f;
    v8h  h2[2];
};

static __device__ inline v8f wmma_f16(v16h a, v16h b, v8f c) {
    // D = A(16x32 f16) * B(32x16 f16) + C(16x16 f32)
    return __builtin_amdgcn_wmma_f32_16x16x32_f16(
        /*neg_a=*/false, a, /*neg_b=*/false, b,
        /*c_mod=*/(short)0, c, /*reuse_a=*/false, /*reuse_b=*/false);
}

// CDNA5 async memory->LDS copy (16 bytes per lane), tracked by ASYNCcnt.
// LDS byte address = low 32 bits of the generic shared pointer (flat->LDS
// mapping truncates to addr[31:0], ISA 10.2).
static __device__ inline void async_ld_b128(const void* lds_ptr, const void* gptr) {
    unsigned           laddr = (unsigned)(unsigned long long)lds_ptr;
    unsigned long long gaddr = (unsigned long long)gptr;
    asm volatile("global_load_async_to_lds_b128 %0, %1, off"
                 :: "v"(laddr), "v"(gaddr) : "memory");
}
static __device__ inline void wait_async0() {
    asm volatile("s_wait_asynccnt 0" ::: "memory");
}

// ---------------------------------------------------------------------------
// fp32 -> f16 conversion (x and weights), n multiple of 4
// ---------------------------------------------------------------------------
__global__ __launch_bounds__(256) void cvt_f32_f16(const float* __restrict__ src,
                                                   _Float16* __restrict__ dst, int n) {
    int i = (blockIdx.x * 256 + threadIdx.x) * 4;
    if (i < n) {
        float4 f = *(const float4*)(src + i);
        dst[i + 0] = (_Float16)f.x;
        dst[i + 1] = (_Float16)f.y;
        dst[i + 2] = (_Float16)f.z;
        dst[i + 3] = (_Float16)f.w;
    }
}

// ---------------------------------------------------------------------------
// Generic GEMM: Y[M,N] = A[M,K] * W[N,K]^T + bias[N]
//   layout 1: f16 head-split   [B,H,S,HD]  (Q/K projections)
//   layout 2: f32 row-major    [M,N]       (final Wo projection -> d_out)
//   layout 3: f16 head-split, d-major [B,H,HD,S]  (V, pre-transposed for attn)
// Block tile 128(M) x 64(N), 8 waves in 4x2, each wave 32x32 (2x2 WMMA tiles).
// Double-buffered LDS tiles filled with async global->LDS copies.
// ---------------------------------------------------------------------------
__global__ __launch_bounds__(256) void gemm_wmma_bias(
    const _Float16* __restrict__ A, const _Float16* __restrict__ W,
    const float* __restrict__ bias, _Float16* __restrict__ outH,
    float* __restrict__ outF, int Mtot, int Ntot, int Ktot, int layout) {

    __shared__ _Float16 As[2][128][40];   // 32 cols + 8 pad
    __shared__ _Float16 Ws[2][64][40];

    const int tid    = threadIdx.x;
    const int wave   = tid >> 5;
    const int lane   = tid & 31;
    const int l      = lane & 15;
    const int hi     = lane >> 4;
    const int wm     = wave >> 1;      // 0..3 -> M offset wm*32
    const int wn     = wave & 1;       // 0..1 -> N offset wn*32
    const int mblock = blockIdx.x * 128;
    const int nblock = blockIdx.y * 64;

    // per-thread cooperative-copy coordinates
    const int ar = tid >> 1, ac = (tid & 1) * 16;   // A tile 128x32, 16 halves/thr
    const int wr = tid >> 2, wc = (tid & 3) * 8;    // W tile 64x32,   8 halves/thr

    v8f acc[2][2];
    acc[0][0] = {}; acc[0][1] = {}; acc[1][0] = {}; acc[1][1] = {};

    const int nk = Ktot / 32;

    // prologue: stage tile 0 into buffer 0
    {
        const _Float16* ag = A + (size_t)(mblock + ar) * Ktot + ac;
        async_ld_b128(&As[0][ar][ac],     ag);
        async_ld_b128(&As[0][ar][ac + 8], ag + 8);
        async_ld_b128(&Ws[0][wr][wc], W + (size_t)(nblock + wr) * Ktot + wc);
    }

    for (int kt = 0; kt < nk; ++kt) {
        const int buf = kt & 1;
        wait_async0();
        __syncthreads();                 // tile kt visible; tile kt-1 fully consumed
        if (kt + 1 < nk) {               // stream tile kt+1 into the other buffer
            const int k1 = (kt + 1) * 32;
            const _Float16* ag = A + (size_t)(mblock + ar) * Ktot + k1 + ac;
            async_ld_b128(&As[buf ^ 1][ar][ac],     ag);
            async_ld_b128(&As[buf ^ 1][ar][ac + 8], ag + 8);
            async_ld_b128(&Ws[buf ^ 1][wr][wc],
                          W + (size_t)(nblock + wr) * Ktot + k1 + wc);
        }

        Frag16 fa[2], fb[2];
#pragma unroll
        for (int t = 0; t < 2; ++t) {
            int arow = wm * 32 + t * 16 + l;          // A row (M)
            fa[t].h2[0] = *(const v8h*)&As[buf][arow][hi * 8];
            fa[t].h2[1] = *(const v8h*)&As[buf][arow][hi * 8 + 16];
            int brow = wn * 32 + t * 16 + l;          // W row (N)
            fb[t].h2[0] = *(const v8h*)&Ws[buf][brow][hi * 8];
            fb[t].h2[1] = *(const v8h*)&Ws[buf][brow][hi * 8 + 16];
        }
#pragma unroll
        for (int mi = 0; mi < 2; ++mi)
#pragma unroll
            for (int ni = 0; ni < 2; ++ni)
                acc[mi][ni] = wmma_f16(fa[mi].f, fb[ni].f, acc[mi][ni]);
    }

    // epilogue: add bias, scatter per layout (C layout: vgpr i -> row, lane -> col)
#pragma unroll
    for (int mi = 0; mi < 2; ++mi) {
#pragma unroll
        for (int ni = 0; ni < 2; ++ni) {
            int gn = nblock + wn * 32 + ni * 16 + l;
            float bv = bias[gn];
            int gm0 = mblock + wm * 32 + mi * 16 + hi * 8;
            if (layout == 3) {
                // V pre-transposed: [B,H,HD,S]; rows gm0..gm0+7 -> contiguous s
                int b = gm0 >> 12, s = gm0 & (S_ - 1);
                int h = gn >> 6, hd = gn & (HD_ - 1);
                v8h pk;
#pragma unroll
                for (int i = 0; i < 8; ++i) pk[i] = (_Float16)(acc[mi][ni][i] + bv);
                *(v8h*)(outH + (((size_t)b * H_ + h) * HD_ + hd) * S_ + s) = pk;
            } else {
#pragma unroll
                for (int i = 0; i < 8; ++i) {
                    int gm = gm0 + i;
                    float v = acc[mi][ni][i] + bv;
                    if (layout == 2) {
                        outF[(size_t)gm * Ntot + gn] = v;
                    } else {  // layout 1
                        int b = gm >> 12, s = gm & (S_ - 1);
                        int h = gn >> 6, hd = gn & (HD_ - 1);
                        outH[(((size_t)b * H_ + h) * S_ + s) * HD_ + hd] = (_Float16)v;
                    }
                }
            }
        }
    }
}

// ---------------------------------------------------------------------------
// Flash attention: one block = 128 query rows of one (b,h); 8 waves x 16 rows.
// K tile [32 keys][64 d] and pre-transposed V tile [64 d][32 keys] are
// double-buffered in LDS and streamed with async global->LDS copies.
// Online softmax in fp32 on WMMA accumulators; P goes through per-wave LDS
// to convert C-layout -> A-layout for the PV WMMA.
// ---------------------------------------------------------------------------
__global__ __launch_bounds__(256) void attn_wmma(
    const _Float16* __restrict__ Qh, const _Float16* __restrict__ Kh,
    const _Float16* __restrict__ Vt, _Float16* __restrict__ attnH) {

    __shared__ _Float16 Ks[2][32][72];     // [key][d]   (64 + 8 pad)
    __shared__ _Float16 Vts[2][64][40];    // [d][key]   (32 + 8 pad)
    __shared__ _Float16 Ps[8][16][40];     // per-wave P tile [qrow][key]

    const int tid  = threadIdx.x;
    const int wave = tid >> 5;
    const int lane = tid & 31;
    const int l    = lane & 15;
    const int hi   = lane >> 4;

    const int bh   = blockIdx.y;                     // 0..B*H-1
    const int q0   = blockIdx.x * 128 + wave * 16;   // this wave's query base
    const size_t baseQ = (size_t)bh * S_ * HD_;      // Q/K: [bh][s][d]
    const size_t baseV = (size_t)bh * HD_ * S_;      // Vt:  [bh][d][s]

    // cooperative-copy coordinates
    const int kr = tid >> 3, kc = (tid & 7) * 8;     // K tile 32x64
    const int vr = tid >> 2, vc = (tid & 3) * 8;     // Vt tile 64x32

    // loop-invariant Q fragments (A operand), d split into two 32-wide K steps
    Frag16 qf[2];
#pragma unroll
    for (int d2 = 0; d2 < 2; ++d2) {
        const _Float16* qp = Qh + baseQ + (size_t)(q0 + l) * HD_ + d2 * 32 + hi * 8;
        qf[d2].h2[0] = *(const v8h*)(qp);
        qf[d2].h2[1] = *(const v8h*)(qp + 16);
    }

    float mrow[8], lrow[8];
    v8f o[4];
#pragma unroll
    for (int i = 0; i < 8; ++i) { mrow[i] = -3.0e38f; lrow[i] = 0.0f; }
    o[0] = {}; o[1] = {}; o[2] = {}; o[3] = {};

    const int nk = S_ / 32;

    // prologue: stage key-tile 0 into buffer 0
    async_ld_b128(&Ks[0][kr][kc],  Kh + baseQ + (size_t)kr * HD_ + kc);
    async_ld_b128(&Vts[0][vr][vc], Vt + baseV + (size_t)vr * S_ + vc);

    for (int kt = 0; kt < nk; ++kt) {
        const int buf = kt & 1;
        wait_async0();
        __syncthreads();                 // tile kt ready; tile kt-1 consumed
        if (kt + 1 < nk) {
            const int k1 = (kt + 1) * 32;
            async_ld_b128(&Ks[buf ^ 1][kr][kc],
                          Kh + baseQ + (size_t)(k1 + kr) * HD_ + kc);
            async_ld_b128(&Vts[buf ^ 1][vr][vc],
                          Vt + baseV + (size_t)vr * S_ + k1 + vc);
        }

        // scores: S = Q * K^T  (two 16-key N tiles)
        v8f s0 = {}, s1 = {};
#pragma unroll
        for (int d2 = 0; d2 < 2; ++d2) {
            Frag16 kb0, kb1;
            kb0.h2[0] = *(const v8h*)&Ks[buf][l][d2 * 32 + hi * 8];
            kb0.h2[1] = *(const v8h*)&Ks[buf][l][d2 * 32 + hi * 8 + 16];
            kb1.h2[0] = *(const v8h*)&Ks[buf][16 + l][d2 * 32 + hi * 8];
            kb1.h2[1] = *(const v8h*)&Ks[buf][16 + l][d2 * 32 + hi * 8 + 16];
            s0 = wmma_f16(qf[d2].f, kb0.f, s0);
            s1 = wmma_f16(qf[d2].f, kb1.f, s1);
        }
        s0 = s0 * SCALE_;
        s1 = s1 * SCALE_;

        // online softmax (rows live one-per-vgpr, cols across 16 lanes)
        float alpha[8];
#pragma unroll
        for (int i = 0; i < 8; ++i) {
            float mx = fmaxf(s0[i], s1[i]);
#pragma unroll
            for (int off = 1; off < 16; off <<= 1)
                mx = fmaxf(mx, __shfl_xor(mx, off, 32));
            float mnew = fmaxf(mrow[i], mx);
            alpha[i] = __expf(mrow[i] - mnew);
            float p0 = __expf(s0[i] - mnew);
            float p1 = __expf(s1[i] - mnew);
            float sum = p0 + p1;
#pragma unroll
            for (int off = 1; off < 16; off <<= 1)
                sum += __shfl_xor(sum, off, 32);
            lrow[i] = lrow[i] * alpha[i] + sum;
            mrow[i] = mnew;
            int pr = hi * 8 + i;
            Ps[wave][pr][l]      = (_Float16)p0;
            Ps[wave][pr][16 + l] = (_Float16)p1;
        }
#pragma unroll
        for (int t = 0; t < 4; ++t)
#pragma unroll
            for (int i = 0; i < 8; ++i) o[t][i] *= alpha[i];

        // wave-local LDS store->load ordering for the P tile
        asm volatile("s_wait_dscnt 0" ::: "memory");

        // P fragment (A layout) and PV WMMAs over HD in 4 N-tiles
        Frag16 pf;
        pf.h2[0] = *(const v8h*)&Ps[wave][l][hi * 8];
        pf.h2[1] = *(const v8h*)&Ps[wave][l][hi * 8 + 16];
#pragma unroll
        for (int t = 0; t < 4; ++t) {
            Frag16 vb;
            vb.h2[0] = *(const v8h*)&Vts[buf][t * 16 + l][hi * 8];
            vb.h2[1] = *(const v8h*)&Vts[buf][t * 16 + l][hi * 8 + 16];
            o[t] = wmma_f16(pf.f, vb.f, o[t]);
        }
    }

    // normalize and store f16 into merged-head [B,S,D] layout
    const int b = bh >> 3;         // bh / H_
    const int h = bh & (H_ - 1);
    float inv[8];
#pragma unroll
    for (int i = 0; i < 8; ++i) inv[i] = 1.0f / lrow[i];
#pragma unroll
    for (int t = 0; t < 4; ++t) {
        int col = h * HD_ + t * 16 + l;
#pragma unroll
        for (int i = 0; i < 8; ++i) {
            int s = q0 + hi * 8 + i;
            attnH[((size_t)(b * S_ + s)) * D_ + col] = (_Float16)(o[t][i] * inv[i]);
        }
    }
}

// ---------------------------------------------------------------------------
// Host launcher
// ---------------------------------------------------------------------------
extern "C" void kernel_launch(void* const* d_in, const int* in_sizes, int n_in,
                              void* d_out, int out_size, void* d_ws, size_t ws_size,
                              hipStream_t stream) {
    const float* x  = (const float*)d_in[0];
    const float* Wq = (const float*)d_in[1];
    const float* bq = (const float*)d_in[2];
    const float* Wk = (const float*)d_in[3];
    const float* bk = (const float*)d_in[4];
    const float* Wv = (const float*)d_in[5];
    const float* bv = (const float*)d_in[6];
    const float* Wo = (const float*)d_in[7];
    const float* bo = (const float*)d_in[8];
    float* out = (float*)d_out;

    // workspace layout (bytes)
    char* ws = (char*)d_ws;
    const size_t XH = (size_t)M_ * D_ * 2;            // 8 MiB
    const size_t WH = (size_t)D_ * D_ * 2;            // 512 KiB
    const size_t QH = (size_t)B_ * H_ * S_ * HD_ * 2; // 8 MiB
    size_t off = 0;
    _Float16* xh  = (_Float16*)(ws + off); off += XH;
    _Float16* wqh = (_Float16*)(ws + off); off += WH;
    _Float16* wkh = (_Float16*)(ws + off); off += WH;
    _Float16* wvh = (_Float16*)(ws + off); off += WH;
    _Float16* woh = (_Float16*)(ws + off); off += WH;
    _Float16* Qh  = (_Float16*)(ws + off); off += QH;
    _Float16* Kh  = (_Float16*)(ws + off); off += QH;
    _Float16* Vth = (_Float16*)(ws + off); off += QH;   // V pre-transposed [B,H,HD,S]
    _Float16* aH  = (_Float16*)(ws + off); off += XH;

    // 1) convert fp32 -> f16
    cvt_f32_f16<<<(M_ * D_) / 1024, 256, 0, stream>>>(x,  xh,  M_ * D_);
    cvt_f32_f16<<<(D_ * D_) / 1024, 256, 0, stream>>>(Wq, wqh, D_ * D_);
    cvt_f32_f16<<<(D_ * D_) / 1024, 256, 0, stream>>>(Wk, wkh, D_ * D_);
    cvt_f32_f16<<<(D_ * D_) / 1024, 256, 0, stream>>>(Wv, wvh, D_ * D_);
    cvt_f32_f16<<<(D_ * D_) / 1024, 256, 0, stream>>>(Wo, woh, D_ * D_);

    // 2) Q/K projections -> [B,H,S,HD]; V -> pre-transposed [B,H,HD,S]
    dim3 ggrid(M_ / 128, D_ / 64);
    gemm_wmma_bias<<<ggrid, 256, 0, stream>>>(xh, wqh, bq, Qh,  nullptr, M_, D_, D_, 1);
    gemm_wmma_bias<<<ggrid, 256, 0, stream>>>(xh, wkh, bk, Kh,  nullptr, M_, D_, D_, 1);
    gemm_wmma_bias<<<ggrid, 256, 0, stream>>>(xh, wvh, bv, Vth, nullptr, M_, D_, D_, 3);

    // 3) flash attention -> merged-head f16 [B,S,D]
    dim3 agrid(S_ / 128, B_ * H_);
    attn_wmma<<<agrid, 256, 0, stream>>>(Qh, Kh, Vth, aH);

    // 4) output projection -> fp32 d_out
    gemm_wmma_bias<<<ggrid, 256, 0, stream>>>(aH, woh, bo, nullptr, out, M_, D_, D_, 2);

    (void)in_sizes; (void)n_in; (void)out_size; (void)ws_size;
}
